// newBasicBlock_76115410419908
// MI455X (gfx1250) — compile-verified, matrix operand order checked by reading, not run on hardware
//
#include <hip/hip_runtime.h>
#include <hip/hip_bf16.h>
#include <stdint.h>

typedef _Float16 h16;
typedef __attribute__((ext_vector_type(16))) _Float16 v16h;
typedef __attribute__((ext_vector_type(4)))  _Float16 v4h;
typedef __attribute__((ext_vector_type(8)))  float    v8f;
typedef __attribute__((ext_vector_type(4)))  float    v4f;

#define KSIZE 36864          // 64*64*9
#define NPIX  1024           // 32*32
#define NIMG  128
#define NCH   64

// LDS layout (dynamic):
//   sB: 340 pixels (10 rows x 34 cols) x 72-padded cin  -> 48,960 B
//   sA: 2 x (64 cout x 72-padded cin)                   -> 18,432 B
#define SB_BYTES 48960
#define SA_BYTES 9216
#define LDS_BYTES (SB_BYTES + 2 * SA_BYTES)

// Async 16-byte copy global -> LDS (gfx1250, ASYNCcnt-tracked).
static __device__ __forceinline__ void async_copy_b128(uint32_t lds_off, const void* gaddr)
{
    asm volatile("global_load_async_to_lds_b128 %0, %1, off"
                 :: "v"(lds_off), "v"(gaddr) : "memory");
}
static __device__ __forceinline__ void wait_async0()
{
    asm volatile("s_wait_asynccnt 0x0" ::: "memory");
}

// ---------------------------------------------------------------------------
// Fold new_param into both conv kernels; store f16 [cout][tap][cin] so the
// GEMM K index is tap*64 + cin (tap-major K) and each row slice is 128 B.
// ---------------------------------------------------------------------------
__global__ __launch_bounds__(256) void prep_weights(
    const float* __restrict__ origin, const float* __restrict__ P,
    const float* __restrict__ np, h16* __restrict__ W1, h16* __restrict__ W2)
{
    __shared__ float a[40];
    if (threadIdx.x < 40) a[threadIdx.x] = np[threadIdx.x];
    __syncthreads();
    for (int f = blockIdx.x * blockDim.x + threadIdx.x; f < KSIZE;
         f += gridDim.x * blockDim.x) {
        int kx   = f % 3;
        int ky   = (f / 3) % 3;
        int cin  = (f / 9) % 64;
        int cout = f / 576;
        float w1 = origin[f];
        float w2 = a[39] * origin[KSIZE + f];   // faithful loop-leak scaling
        #pragma unroll 8
        for (int i = 0; i < 40; ++i) {
            w1 += a[i] * P[i * (2 * KSIZE) + f];
            w2 += a[i] * P[i * (2 * KSIZE) + KSIZE + f];
        }
        int dst = cout * 576 + (ky * 3 + kx) * 64 + cin;
        W1[dst] = (h16)w1;
        W2[dst] = (h16)w2;
    }
}

// NCHW f32 -> NHWC f16 (so a pixel's 64 channels are 128 contiguous bytes).
__global__ __launch_bounds__(256) void x_to_nhwc_f16(
    const float* __restrict__ X, h16* __restrict__ Xh)
{
    const int total = NIMG * NPIX * NCH;
    for (int j = blockIdx.x * blockDim.x + threadIdx.x; j < total;
         j += gridDim.x * blockDim.x) {
        int c  = j & 63;
        int p  = (j >> 6) & (NPIX - 1);
        int nn = j >> 16;
        Xh[j] = (h16)X[(nn * NCH + c) * NPIX + p];
    }
}

// Assemble a v16h WMMA fragment from LDS (16-bit A/B lane layout):
// element e<8 -> K = 8*half + e ; e>=8 -> K = 16 + 8*half + (e-8).
static __device__ __forceinline__ v16h load_frag(const h16* p)
{
    v4h q0 = *(const v4h*)(p);
    v4h q1 = *(const v4h*)(p + 4);
    v4h q2 = *(const v4h*)(p + 16);
    v4h q3 = *(const v4h*)(p + 20);
    v16h r;
    r[0]=q0[0];  r[1]=q0[1];  r[2]=q0[2];  r[3]=q0[3];
    r[4]=q1[0];  r[5]=q1[1];  r[6]=q1[2];  r[7]=q1[3];
    r[8]=q2[0];  r[9]=q2[1];  r[10]=q2[2]; r[11]=q2[3];
    r[12]=q3[0]; r[13]=q3[1]; r[14]=q3[2]; r[15]=q3[3];
    return r;
}

// ---------------------------------------------------------------------------
// Implicit-GEMM 3x3 conv, stride 1, pad 1. Input NHWC f16, output NCHW f32.
// Block = one image x 256 pixels (8 rows). C[64 x 256] = A[64x576]*B[576x256],
// K tap-major, 18 wmma K-steps. 8 waves: wave owns 32(M) x 64(N) -> 8 acc.
// B tile staged once via async-to-LDS b128; per-tap A slice async + double-
// buffered (prefetch tap+1 during tap's compute).
// ---------------------------------------------------------------------------
__global__ __launch_bounds__(256) void conv3x3_wmma(
    const h16* __restrict__ Xh,     // [128][32][32][64]
    const h16* __restrict__ Wf,     // [64][9][64]
    float* __restrict__ Y)          // [128][64][32][32]
{
    extern __shared__ __align__(16) char smem[];
    h16* sB = (h16*)smem;                       // [340][72]
    h16* sA = (h16*)(smem + SB_BYTES);          // [2][64][72]

    const int n     = blockIdx.y;
    const int chunk = blockIdx.x;               // 8 output rows per chunk
    const int tid   = threadIdx.x;
    const int wave  = tid >> 5;
    const int lane  = tid & 31;
    const int half  = lane >> 4;
    const int l16   = lane & 15;
    const int m0    = (wave & 1) * 32;          // cout base (2 strips of 16)
    const int n0    = (wave >> 1) * 64;         // pixel base
    const int oh0   = chunk * 8;

    const uint32_t sB_off = (uint32_t)(uintptr_t)sB;   // LDS byte offset
    const uint32_t sA_off = (uint32_t)(uintptr_t)sA;

    // ---- Stage B tile: rows oh0-1..oh0+8, cols -1..32, 64 ch (async b128).
    for (int idx = tid; idx < 340 * 8; idx += 256) {
        int g   = idx & 7;
        int pix = idx >> 3;
        int r   = pix / 34;
        int c   = pix % 34;
        int hh  = oh0 - 1 + r;
        int ww  = c - 1;
        if (hh >= 0 && hh < 32 && ww >= 0 && ww < 32) {
            const h16* gp = Xh + (size_t)(((n * 32 + hh) * 32 + ww) * 64 + g * 8);
            async_copy_b128(sB_off + (uint32_t)(pix * 144 + g * 16), gp);
        } else {
            v4f z = {0.0f, 0.0f, 0.0f, 0.0f};
            *(v4f*)(smem + (pix * 144 + g * 16)) = z;   // ds_store_b128 zero
        }
    }
    // ---- Stage A slice for tap 0 into buffer 0 (async b128).
    for (int idx = tid; idx < 64 * 8; idx += 256) {
        int g = idx & 7, m = idx >> 3;
        async_copy_b128(sA_off + (uint32_t)(m * 144 + g * 16),
                        Wf + (size_t)(m * 576 + 0 * 64 + g * 8));
    }
    wait_async0();
    __syncthreads();

    v8f acc[2][4];
    #pragma unroll
    for (int j = 0; j < 2; ++j)
        #pragma unroll
        for (int s = 0; s < 4; ++s)
            #pragma unroll
            for (int e = 0; e < 8; ++e) acc[j][s][e] = 0.0f;

    for (int tap = 0; tap < 9; ++tap) {
        const int buf = tap & 1;
        // Prefetch next tap's weights into the other buffer (overlapped).
        if (tap < 8) {
            const uint32_t dst = sA_off + (uint32_t)((buf ^ 1) * SA_BYTES);
            for (int idx = tid; idx < 64 * 8; idx += 256) {
                int g = idx & 7, m = idx >> 3;
                async_copy_b128(dst + (uint32_t)(m * 144 + g * 16),
                                Wf + (size_t)(m * 576 + (tap + 1) * 64 + g * 8));
            }
        }
        const int dy = tap / 3, dx = tap % 3;
        const h16* sAb = sA + buf * (SA_BYTES / 2 /*halves*/) * 0 + buf * (64 * 72);

        #pragma unroll
        for (int kc = 0; kc < 2; ++kc) {
            const int kb = kc * 32 + half * 8;
            v16h a0 = load_frag(&sAb[(m0 + l16) * 72 + kb]);
            v16h a1 = load_frag(&sAb[(m0 + 16 + l16) * 72 + kb]);
            #pragma unroll
            for (int s = 0; s < 4; ++s) {
                const int p  = n0 + s * 16 + l16;
                const int br = (p >> 5) + dy;
                const int bc = (p & 31) + dx;
                v16h b = load_frag(&sB[(br * 34 + bc) * 72 + kb]);
                acc[0][s] = __builtin_amdgcn_wmma_f32_16x16x32_f16(
                    false, a0, false, b, (short)0, acc[0][s], false, false);
                acc[1][s] = __builtin_amdgcn_wmma_f32_16x16x32_f16(
                    false, a1, false, b, (short)0, acc[1][s], false, false);
            }
        }
        wait_async0();       // next tap's A slice landed
        __syncthreads();     // whole workgroup sees it / done reading old buf
    }

    // D layout: VGPR rr -> M = base + rr + 8*half ; N = pixel n0 + s*16 + l16.
    #pragma unroll
    for (int j = 0; j < 2; ++j)
        #pragma unroll
        for (int s = 0; s < 4; ++s) {
            const int pg = chunk * 256 + n0 + s * 16 + l16;
            #pragma unroll
            for (int rr = 0; rr < 8; ++rr) {
                const int m = m0 + 16 * j + rr + 8 * half;
                Y[(size_t)(n * NCH + m) * NPIX + pg] = acc[j][s][rr];
            }
        }
}

// ---------------------------------------------------------------------------
// Per-channel batch statistics: one block per channel, tree reduction.
// ---------------------------------------------------------------------------
__global__ __launch_bounds__(256) void bn_stats(
    const float* __restrict__ Y, float* __restrict__ mean, float* __restrict__ invstd)
{
    const int c = blockIdx.x;
    float s = 0.0f, ss = 0.0f;
    for (int idx = threadIdx.x; idx < NIMG * NPIX; idx += 256) {
        int nn = idx >> 10;
        int p  = idx & (NPIX - 1);
        float v = Y[(nn * NCH + c) * NPIX + p];
        s += v; ss += v * v;
    }
    __shared__ float rs[256], rq[256];
    rs[threadIdx.x] = s; rq[threadIdx.x] = ss;
    __syncthreads();
    for (int off = 128; off > 0; off >>= 1) {
        if (threadIdx.x < off) {
            rs[threadIdx.x] += rs[threadIdx.x + off];
            rq[threadIdx.x] += rq[threadIdx.x + off];
        }
        __syncthreads();
    }
    if (threadIdx.x == 0) {
        const float inv_n = 1.0f / (float)(NIMG * NPIX);
        float m  = rs[0] * inv_n;
        float vr = rq[0] * inv_n - m * m;
        mean[c]   = m;
        invstd[c] = rsqrtf(vr + 1e-5f);
    }
}

// BN + ReLU -> NHWC f16 activations for the layer-2 GEMM.
__global__ __launch_bounds__(256) void bn_act_f16(
    const float* __restrict__ Y, const float* __restrict__ mean,
    const float* __restrict__ invstd, const float* __restrict__ gamma,
    const float* __restrict__ beta, h16* __restrict__ A)
{
    const int total = NIMG * NPIX * NCH;
    for (int j = blockIdx.x * blockDim.x + threadIdx.x; j < total;
         j += gridDim.x * blockDim.x) {
        int c  = j & 63;
        int p  = (j >> 6) & (NPIX - 1);
        int nn = j >> 16;
        float v = (Y[(nn * NCH + c) * NPIX + p] - mean[c]) * invstd[c] * gamma[c] + beta[c];
        A[j] = (h16)fmaxf(v, 0.0f);
    }
}

// BN2 + identity shortcut + ReLU -> f32 output (NCHW).
__global__ __launch_bounds__(256) void bn_add_relu(
    const float* __restrict__ Y, const float* __restrict__ mean,
    const float* __restrict__ invstd, const float* __restrict__ gamma,
    const float* __restrict__ beta, const float* __restrict__ X,
    float* __restrict__ Out)
{
    const int total = NIMG * NCH * NPIX;
    for (int i = blockIdx.x * blockDim.x + threadIdx.x; i < total;
         i += gridDim.x * blockDim.x) {
        int c = (i >> 10) & (NCH - 1);
        float v = (Y[i] - mean[c]) * invstd[c] * gamma[c] + beta[c] + X[i];
        Out[i] = fmaxf(v, 0.0f);
    }
}

// ---------------------------------------------------------------------------
extern "C" void kernel_launch(void* const* d_in, const int* in_sizes, int n_in,
                              void* d_out, int out_size, void* d_ws, size_t ws_size,
                              hipStream_t stream)
{
    const float* x      = (const float*)d_in[0];
    const float* origin = (const float*)d_in[1];
    const float* P      = (const float*)d_in[2];
    const float* np     = (const float*)d_in[3];
    const float* g1     = (const float*)d_in[4];
    const float* b1     = (const float*)d_in[5];
    const float* g2     = (const float*)d_in[6];
    const float* b2     = (const float*)d_in[7];
    float* out = (float*)d_out;

    char* ws = (char*)d_ws;
    h16*   W1    = (h16*)(ws);                          // 73,728 B
    h16*   W2    = (h16*)(ws + 73728);                  // 73,728 B
    h16*   xh    = (h16*)(ws + 147456);                 // 16,777,216 B (NHWC f16)
    float* craw  = (float*)(ws + 16924672);             // 33,554,432 B (reused)
    h16*   act   = (h16*)(ws + 50479104);               // 16,777,216 B (NHWC f16)
    float* stats = (float*)(ws + 67256320);             // 4 x 64 f32
    float* mean1 = stats,       *istd1 = stats + 64;
    float* mean2 = stats + 128, *istd2 = stats + 192;

    prep_weights<<<144, 256, 0, stream>>>(origin, P, np, W1, W2);
    x_to_nhwc_f16<<<4096, 256, 0, stream>>>(x, xh);

    dim3 grid(4, 128);
    conv3x3_wmma<<<grid, 256, LDS_BYTES, stream>>>(xh, W1, craw);
    bn_stats<<<64, 256, 0, stream>>>(craw, mean1, istd1);
    bn_act_f16<<<8192, 256, 0, stream>>>(craw, mean1, istd1, g1, b1, act);

    conv3x3_wmma<<<grid, 256, LDS_BYTES, stream>>>(act, W2, craw);
    bn_stats<<<64, 256, 0, stream>>>(craw, mean2, istd2);
    bn_add_relu<<<8192, 256, 0, stream>>>(craw, mean2, istd2, g2, b2, x, out);
}
